// Dynamic_MLP_A_58634893525052
// MI455X (gfx1250) — matrix operand order repaired and checked by reading, not compiled, not run on hardware
//
#include <hip/hip_runtime.h>

typedef _Float16 h16;
typedef _Float16 v16h  __attribute__((ext_vector_type(16)));
typedef _Float16 v8h   __attribute__((ext_vector_type(8)));
typedef float    v8f   __attribute__((ext_vector_type(8)));
typedef unsigned int u32x4 __attribute__((ext_vector_type(4)));
typedef int      i32x8 __attribute__((ext_vector_type(8)));
typedef int      i32x4 __attribute__((ext_vector_type(4)));

#define JOIN16(a,b) __builtin_shufflevector(a, b, 0,1,2,3,4,5,6,7,8,9,10,11,12,13,14,15)

// ---------------- geometry ----------------
// tokens T = 4*4096 = 16384, inplanes = planes = locplanes = 128
#define TOK_PER_WG 64
#define NI        128          // outer K loop over i
#define LOCP      136          // padded half stride for 128-wide rows (+8 halfs = 16B)
#define WROW      136          // W block LDS row stride in halfs (TDM pads 4 dwords/row)
// LDS byte offsets (single __shared__ blob, assumed to start at LDS offset 0)
#define OFF_WBUF0 0
#define OFF_WBUF1 34816        // 128*272
#define OFF_LOC   69632
#define OFF_IMG   87040
#define OFF_BIAS  104448
#define SMEM_TOTAL 139264
// epilogue overlays the (dead) W buffers
#define OFF_Y     0            // 64 tokens * 129 floats (padded) = 33024 B
#define OFF_MU    33024
#define OFF_RS    33280

// ---------------- TDM issue: one 128x128 f16 tile of W2 into LDS ----------------
__device__ __forceinline__ void tdm_issue_wblock(const h16* wbase, int iblk, unsigned lds_off) {
  unsigned long long ga = (unsigned long long)wbase + (unsigned long long)iblk * 32768ull; // 128*128*2B
  u32x4 g0;
  g0[0] = 1u;                                    // count=1, user mode
  g0[1] = lds_off;                               // lds_addr [63:32]
  g0[2] = (unsigned)ga;                          // global_addr [95:64]
  g0[3] = (unsigned)((ga >> 32) & 0x1FFFFFFu) | (2u << 30); // addr[56:32] | type=2
  i32x8 g1;
  g1[0] = (1 << 16) | (1 << 20) | (5 << 22) | (3 << 25);
  //        data=2B   pad_en     every 64dw   pad 4dw  (row 256B -> stride 272B in LDS)
  g1[1] = (128 << 16);   // tensor_dim0 = 128 (bits 79:48, low half here)
  g1[2] = (128 << 16);   // tensor_dim0 hi = 0 ; tensor_dim1 = 128 (low half)
  g1[3] = (128 << 16);   // tensor_dim1 hi = 0 ; tile_dim0 = 128
  g1[4] = 128;           // tile_dim1 = 128 ; tile_dim2 = 0
  g1[5] = 128;           // tensor_dim0_stride = 128 (elements)
  g1[6] = 0;
  g1[7] = 0;
  i32x4 gz = {0, 0, 0, 0};
#if defined(__clang_major__) && (__clang_major__ >= 23)
  i32x8 gz8 = {0, 0, 0, 0, 0, 0, 0, 0};
  __builtin_amdgcn_tensor_load_to_lds(g0, g1, gz, gz, gz8, 0);
#else
  __builtin_amdgcn_tensor_load_to_lds(g0, g1, gz, gz, 0);
#endif
}

// ---------------- preprocessing ----------------
__global__ void cvtW_kernel(const float* __restrict__ W, h16* __restrict__ Wh) {
  size_t i = (size_t)blockIdx.x * 256 + threadIdx.x;   // 16384*128 elems
  Wh[i] = (h16)W[i];
}
__global__ void cvtB_kernel(const float* __restrict__ b, h16* __restrict__ bT) {
  int idx = blockIdx.x * 256 + threadIdx.x;            // 128*128 elems
  int o = idx >> 7, i = idx & 127;
  bT[idx] = (h16)b[i * 128 + o];                       // bT[o][i]
}

// ---------------- main fused kernel ----------------
__global__ __launch_bounds__(256) void dynmlp_kernel(
    const float* __restrict__ img, const float* __restrict__ loc,
    const h16* __restrict__ Wh, const h16* __restrict__ bT,
    const float* __restrict__ gamma, const float* __restrict__ beta,
    float* __restrict__ out)
{
  __shared__ __align__(16) unsigned char smem[SMEM_TOTAL];
  h16* loc_s  = (h16*)(smem + OFF_LOC);
  h16* img_s  = (h16*)(smem + OFF_IMG);
  h16* bias_s = (h16*)(smem + OFF_BIAS);

  const int tid  = threadIdx.x;
  const int tok0 = blockIdx.x * TOK_PER_WG;

  // kick off TDM load of W block 0 early (wave 0)
  if (tid == 0) tdm_issue_wblock(Wh, 0, OFF_WBUF0);

  // stage loc/img tiles (f32 -> f16, padded rows)
  for (int idx = tid; idx < TOK_PER_WG * 128; idx += 256) {
    int t = idx >> 7, c = idx & 127;
    loc_s[t * LOCP + c] = (h16)loc[(size_t)(tok0 + t) * 128 + c];
    img_s[t * LOCP + c] = (h16)img[(size_t)(tok0 + t) * 128 + c];
  }
  // stage transposed bias matrix bT[o][i]
  for (int idx = tid; idx < 128 * 128; idx += 256) {
    int o = idx >> 7, i = idx & 127;
    bias_s[o * WROW + i] = bT[idx];
  }
  __syncthreads();  // tiles visible

  const int lane  = tid & 31;
  const int wave  = tid >> 5;
  const int n     = lane & 15;
  const int hi    = lane >> 4;
  const int mtile = wave >> 1;         // 4 M-tiles of 16 tokens
  const int ohalf = wave & 1;          // 2 waves split the 128 outputs
  const int t     = mtile * 16 + n;    // this lane's token row
  const int obase = ohalf * 64;

  // preload this lane's loc fragments (A-layout: lo lanes K{0-7,16-23}, hi lanes K{8-15,24-31})
  v16h locf[4];
  {
    const h16* lp0 = loc_s + t * LOCP;
#pragma unroll
    for (int cc = 0; cc < 4; ++cc) {
      const h16* lp = lp0 + cc * 32;
      v8h a0 = *(const v8h*)(lp + hi * 8);
      v8h a1 = *(const v8h*)(lp + 16 + hi * 8);
      locf[cc] = JOIN16(a0, a1);
    }
  }

  v8f acc[4] = {v8f{}, v8f{}, v8f{}, v8f{}};

  for (int i = 0; i < NI; ++i) {
    if (tid == 0 && (i + 1) < NI)
      tdm_issue_wblock(Wh, i + 1, ((i + 1) & 1) ? OFF_WBUF1 : OFF_WBUF0);
    if (wave == 0) {
      if ((i + 1) < NI) __builtin_amdgcn_s_wait_tensorcnt(1);  // block i landed
      else              __builtin_amdgcn_s_wait_tensorcnt(0);
    }
    __syncthreads();  // all waves: block i ready

    const h16* wb = (const h16*)(smem + ((i & 1) ? OFF_WBUF1 : OFF_WBUF0));
    h16 iv = img_s[t * LOCP + i];      // img[t,i] scalar per lane

#pragma unroll
    for (int cc = 0; cc < 4; ++cc) {
      v16h a = locf[cc] * iv;          // A = img[t,i] * loc[t, c-chunk] (v_pk_mul_f16)
#pragma unroll
      for (int ot = 0; ot < 4; ++ot) {
        const h16* bp = wb + (obase + ot * 16 + n) * WROW + cc * 32 + hi * 8;
        v8h b0 = *(const v8h*)bp;
        v8h b1 = *(const v8h*)(bp + 16);
        v16h bf = JOIN16(b0, b1);
        acc[ot] = __builtin_amdgcn_wmma_f32_16x16x32_f16(
            false, a, false, bf, (short)0, acc[ot], false, false);
      }
    }
    __syncthreads();  // block i fully consumed before buffer reuse
  }

  // fold bias: y += img[t,:] @ bT  (K = 128 over i)
#pragma unroll
  for (int kc = 0; kc < 4; ++kc) {
    const h16* ip = img_s + t * LOCP + kc * 32;
    v8h a0 = *(const v8h*)(ip + hi * 8);
    v8h a1 = *(const v8h*)(ip + 16 + hi * 8);
    v16h a = JOIN16(a0, a1);
#pragma unroll
    for (int ot = 0; ot < 4; ++ot) {
      const h16* bp = bias_s + (obase + ot * 16 + n) * WROW + kc * 32 + hi * 8;
      v8h b0 = *(const v8h*)bp;
      v8h b1 = *(const v8h*)(bp + 16);
      v16h bf = JOIN16(b0, b1);
      acc[ot] = __builtin_amdgcn_wmma_f32_16x16x32_f16(
          false, a, false, bf, (short)0, acc[ot], false, false);
    }
  }

  // spill accumulators to padded LDS (C-layout: VGPR r -> M = r + 8*hi, N = n)
  float* y_s = (float*)(smem + OFF_Y);
#pragma unroll
  for (int ot = 0; ot < 4; ++ot) {
#pragma unroll
    for (int r = 0; r < 8; ++r) {
      y_s[(mtile * 16 + 8 * hi + r) * 129 + obase + ot * 16 + n] = acc[ot][r];
    }
  }
  __syncthreads();

  // LayerNorm stats: one thread per token (padded rows -> bank-conflict-free)
  float* mu_s = (float*)(smem + OFF_MU);
  float* rs_s = (float*)(smem + OFF_RS);
  if (tid < TOK_PER_WG) {
    const float* yr = y_s + tid * 129;
    float s = 0.f;
    for (int o = 0; o < 128; ++o) s += yr[o];
    float mu = s * (1.f / 128.f);
    float v = 0.f;
    for (int o = 0; o < 128; ++o) { float d = yr[o] - mu; v += d * d; }
    mu_s[tid] = mu;
    rs_s[tid] = rsqrtf(v * (1.f / 128.f) + 1e-5f);
  }
  __syncthreads();

  // normalize + affine + ReLU, coalesced f32 stores
  for (int idx = tid; idx < TOK_PER_WG * 128; idx += 256) {
    int tt = idx >> 7, o = idx & 127;
    float v = (y_s[tt * 129 + o] - mu_s[tt]) * rs_s[tt] * gamma[o] + beta[o];
    out[(size_t)(tok0 + tt) * 128 + o] = v > 0.f ? v : 0.f;
  }
}

extern "C" void kernel_launch(void* const* d_in, const int* in_sizes, int n_in,
                              void* d_out, int out_size, void* d_ws, size_t ws_size,
                              hipStream_t stream) {
  const float* img   = (const float*)d_in[0];
  const float* loc   = (const float*)d_in[1];
  const float* W     = (const float*)d_in[2];
  const float* b     = (const float*)d_in[3];
  const float* gamma = (const float*)d_in[4];
  const float* beta  = (const float*)d_in[5];

  h16* Wh = (h16*)d_ws;                                   // 16384x128 f16 = 4 MB
  h16* bT = (h16*)((char*)d_ws + (size_t)16384 * 128 * 2); // 128x128 f16 = 32 KB

  cvtW_kernel<<<8192, 256, 0, stream>>>(W, Wh);
  cvtB_kernel<<<64,   256, 0, stream>>>(b, bT);
  dynmlp_kernel<<<256, 256, 0, stream>>>(img, loc, Wh, bT, gamma, beta, (float*)d_out);
}
